// MT5DecoderCrossTransformers_6777458393622
// MI455X (gfx1250) — matrix operand order repaired and checked by reading, not compiled, not run on hardware
//
#include <hip/hip_runtime.h>

// ---------------------------------------------------------------------------
// MT5 decoder cross-attention for MI455X (gfx1250), f16 WMMA pipeline with
// async-to-LDS double-buffered K/V staging in the attention kernel.
// B=4, SQ=SK=1024, D=1024, H=16, DH=64
// ---------------------------------------------------------------------------

typedef __attribute__((ext_vector_type(16))) _Float16 v16h;
typedef __attribute__((ext_vector_type(8)))  _Float16 v8h;
typedef __attribute__((ext_vector_type(4)))  _Float16 v4h;
typedef __attribute__((ext_vector_type(8)))  float    v8f;

#define SQ_   1024
#define SK_   1024
#define D_    1024
#define H_    16
#define DH_   64
#define B_    4

__device__ __forceinline__ v8f wmma16(v16h a, v16h b, v8f c) {
    return __builtin_amdgcn_wmma_f32_16x16x32_f16(
        /*neg_a=*/false, a, /*neg_b=*/false, b,
        /*c_mod=*/(short)0, c, /*reuse_a=*/false, /*reuse_b=*/false);
}

// A-fragment (16x32, MxK): lane = (m = lane&15, half = lane>>4)
// elements 0..7  -> K = half*8 + 0..7 ; elements 8..15 -> K = 16 + half*8 + 0..7
__device__ __forceinline__ v16h load_a(const _Float16* base, int ld, int lane) {
    const int r = lane & 15, hh = lane >> 4;
    const _Float16* p = base + (size_t)r * ld + hh * 8;
    union { v16h v; v8h h[2]; } u;
    u.h[0] = *(const v8h*)(p);
    u.h[1] = *(const v8h*)(p + 16);
    return u.v;
}

// B-fragment (32x16, KxN) from N-major storage Bt[n][k] with leading dim ld:
// lane = (n = lane&15, half = lane>>4); elements 0..7 -> K = half*8 + 0..7 ; elements 8..15 -> K = 16 + half*8 + 0..7
__device__ __forceinline__ v16h load_b(const _Float16* base, int ld, int lane) {
    const int n = lane & 15, hh = lane >> 4;
    const _Float16* p = base + (size_t)n * ld + hh * 8;
    union { v16h v; v8h h[2]; } u;
    u.h[0] = *(const v8h*)(p);
    u.h[1] = *(const v8h*)(p + 16);
    return u.v;
}

// Per-lane 16B async copy global -> LDS (ASYNCcnt-tracked, CDNA5).
// LDS byte offset = low 32 bits of the generic pointer (flat LDS aperture).
__device__ __forceinline__ void async_copy_b128(const _Float16* lds_dst,
                                                const _Float16* gsrc) {
    const unsigned l = (unsigned)(unsigned long long)lds_dst;
    asm volatile("global_load_async_to_lds_b128 %0, %1, off"
                 :: "v"(l), "v"(gsrc) : "memory");
}

// ---------------------------------------------------------------------------
// fp32 -> f16 elementwise (vectorized x4)
// ---------------------------------------------------------------------------
__global__ __launch_bounds__(256) void cvt_f32_f16(const float* __restrict__ src,
                                                   _Float16* __restrict__ dst,
                                                   int n4) {
    int i = blockIdx.x * blockDim.x + threadIdx.x;
    if (i < n4) {
        float4 f = ((const float4*)src)[i];
        v4h h;
        h[0] = (_Float16)f.x; h[1] = (_Float16)f.y;
        h[2] = (_Float16)f.z; h[3] = (_Float16)f.w;
        ((v4h*)dst)[i] = h;
    }
}

// ---------------------------------------------------------------------------
// fp32 [R,C] -> f16 [C,R] (weight transpose for N-major B operands)
// ---------------------------------------------------------------------------
__global__ __launch_bounds__(256) void transpose_cvt(const float* __restrict__ src,
                                                     _Float16* __restrict__ dst,
                                                     int R, int C) {
    __shared__ float tile[64][65];
    const int c0 = blockIdx.x * 64, r0 = blockIdx.y * 64;
    const int tx = threadIdx.x & 63, ty = threadIdx.x >> 6;  // ty in 0..3
#pragma unroll
    for (int i = 0; i < 64; i += 4)
        tile[ty + i][tx] = src[(size_t)(r0 + ty + i) * C + c0 + tx];
    __syncthreads();
#pragma unroll
    for (int i = 0; i < 64; i += 4)
        dst[(size_t)(c0 + ty + i) * R + r0 + tx] = (_Float16)tile[tx][ty + i];
}

// ---------------------------------------------------------------------------
// Tiled f16 WMMA GEMM:  C[M,N] = A[M,K] * Bt[N,K]^T, fp32 accumulate.
// Block = 128 threads (4 waves), block tile 128x128, wave tile 64x64.
// EPI 0: f16 out, permuted  [B,H,S,DH]    (q, k projections)
// EPI 1: fp32 out, row-major [M,N]        (final output)
// EPI 2: f16 out, V-transposed [B,H,DH,S] (v projection)
// ---------------------------------------------------------------------------
template <int EPI>
__global__ __launch_bounds__(128) void gemm_f16(const _Float16* __restrict__ A,
                                                const _Float16* __restrict__ Bt,
                                                void* __restrict__ Cout,
                                                int M, int N, int K) {
    const int lane = threadIdx.x & 31;
    const int wave = threadIdx.x >> 5;
    const int row0 = blockIdx.y * 128 + (wave >> 1) * 64;
    const int col0 = blockIdx.x * 128 + (wave & 1) * 64;

    v8f acc[4][4] = {};
    for (int k0 = 0; k0 < K; k0 += 32) {
        v16h a[4], b[4];
#pragma unroll
        for (int i = 0; i < 4; ++i)
            a[i] = load_a(A + (size_t)(row0 + 16 * i) * K + k0, K, lane);
#pragma unroll
        for (int j = 0; j < 4; ++j)
            b[j] = load_b(Bt + (size_t)(col0 + 16 * j) * K + k0, K, lane);
#pragma unroll
        for (int i = 0; i < 4; ++i)
#pragma unroll
            for (int j = 0; j < 4; ++j)
                acc[i][j] = wmma16(a[i], b[j], acc[i][j]);
    }

    const int rr = lane >> 4, cc = lane & 15;
#pragma unroll
    for (int i = 0; i < 4; ++i)
#pragma unroll
        for (int j = 0; j < 4; ++j)
#pragma unroll
            for (int r = 0; r < 8; ++r) {
                const int m = row0 + 16 * i + r + 8 * rr;
                const int n = col0 + 16 * j + cc;
                const float v = acc[i][j][r];
                if (EPI == 1) {
                    ((float*)Cout)[(size_t)m * N + n] = v;
                } else {
                    const int b_ = m >> 10, s = m & (SQ_ - 1);  // m = b*1024 + s
                    const int h_ = n >> 6,  d = n & (DH_ - 1);  // n = h*64 + d
                    size_t idx;
                    if (EPI == 0)
                        idx = (((size_t)(b_ * H_ + h_)) * SQ_ + s) * DH_ + d;
                    else  // EPI == 2: V transposed
                        idx = (((size_t)(b_ * H_ + h_)) * DH_ + d) * SK_ + s;
                    ((_Float16*)Cout)[idx] = (_Float16)v;
                }
            }
}

// ---------------------------------------------------------------------------
// Flash attention, T5 style (no 1/sqrt(dh), additive bias).
// Grid: B*H*(SQ/64) blocks, 128 threads (4 waves); wave owns 16 query rows.
// K/V tiles staged into double-buffered LDS with global_load_async_to_lds_b128
// and shared by all 4 waves (4x L2 traffic reduction + prefetch overlap).
// q : [B,H,SQ,DH] f16 ; k : [B,H,SK,DH] f16 ; vt: [B,H,DH,SK] f16
// bias: [1,H,SQ,SK] fp32 ; ctx: [B,SQ,H*DH] f16
// ---------------------------------------------------------------------------
__global__ __launch_bounds__(128) void attn_kernel(const _Float16* __restrict__ Q,
                                                   const _Float16* __restrict__ Km,
                                                   const _Float16* __restrict__ Vt,
                                                   const float* __restrict__ bias,
                                                   _Float16* __restrict__ ctx) {
    const int tid  = threadIdx.x;
    const int lane = tid & 31;
    const int wave = tid >> 5;
    const int qb = blockIdx.x & 15;        // SQ/64 = 16 blocks per head
    const int bh = blockIdx.x >> 4;        // 0..63
    const int b  = bh >> 4;                // H = 16
    const int h  = bh & (H_ - 1);
    const int q0 = qb * 64 + wave * 16;    // this wave's first query row

    const _Float16* Qp = Q  + ((size_t)bh * SQ_ + q0) * DH_;
    const _Float16* Kp = Km + (size_t)bh * SK_ * DH_;
    const _Float16* Vp = Vt + (size_t)bh * DH_ * SK_;
    const float*    Bp = bias + ((size_t)h * SQ_ + q0) * SK_;

    __shared__ _Float16 Kt[2][32 * 64];    // [sk][dh] tile, double buffered (2x4KB)
    __shared__ _Float16 Vs[2][64 * 32];    // [dh][sk] tile, double buffered (2x4KB)
    __shared__ _Float16 lp[4][16 * 32];    // per-wave P staging (C-layout -> A-layout)
    _Float16* myP = lp[wave];

    // Cooperative async stage of one 32-wide K and Vt tile (4 async ops/lane).
    auto stage = [&](int buf, int sk0) {
#pragma unroll
        for (int c = tid; c < 256; c += 128) {         // K tile: 32 rows x 8 chunks
            const int row = c >> 3, col = (c & 7) * 8;
            async_copy_b128(&Kt[buf][row * 64 + col],
                            Kp + (size_t)(sk0 + row) * DH_ + col);
        }
#pragma unroll
        for (int c = tid; c < 256; c += 128) {         // Vt tile: 64 rows x 4 chunks
            const int row = c >> 2, col = (c & 3) * 8;
            async_copy_b128(&Vs[buf][row * 32 + col],
                            Vp + (size_t)row * SK_ + sk0 + col);
        }
    };

    const v16h aq0 = load_a(Qp, DH_, lane);       // dh 0..31
    const v16h aq1 = load_a(Qp + 32, DH_, lane);  // dh 32..63

    v8f acc[4] = {};
    float m_run[8], l_run[8];
#pragma unroll
    for (int r = 0; r < 8; ++r) { m_run[r] = -1e30f; l_run[r] = 0.0f; }

    const int rr = lane >> 4, cc = lane & 15;

    stage(0, 0);  // prologue: tile 0 in flight (ASYNCcnt = 4)

    for (int sk0 = 0; sk0 < SK_; sk0 += 32) {
        const int buf = (sk0 >> 5) & 1;
        if (sk0 + 32 < SK_) {
            stage(buf ^ 1, sk0 + 32);                       // prefetch next tile
            asm volatile("s_wait_asynccnt 0x4" ::: "memory");  // wait current only
        } else {
            asm volatile("s_wait_asynccnt 0x0" ::: "memory");
        }
        __syncthreads();   // all waves' async copies for this tile visible

        const _Float16* Kb = Kt[buf];
        const _Float16* Vb = Vs[buf];

        // ---- S = Q * K^T (two 16-col tiles), K fragments from LDS ----
        const v16h bk0a = load_b(Kb,                DH_, lane);  // dh 0..31
        const v16h bk0b = load_b(Kb + 32,           DH_, lane);  // dh 32..63
        const v16h bk1a = load_b(Kb + 16 * DH_,     DH_, lane);
        const v16h bk1b = load_b(Kb + 16 * DH_ + 32, DH_, lane);
        v8f s0 = {}, s1 = {};
        s0 = wmma16(aq0, bk0a, s0); s0 = wmma16(aq1, bk0b, s0);
        s1 = wmma16(aq0, bk1a, s1); s1 = wmma16(aq1, bk1b, s1);

        // ---- additive position bias ----
#pragma unroll
        for (int r = 0; r < 8; ++r) {
            const float* brow = Bp + (size_t)(r + 8 * rr) * SK_ + sk0 + cc;
            s0[r] += brow[0];
            s1[r] += brow[16];
        }

        // ---- online softmax (row state replicated across 16-lane half) ----
#pragma unroll
        for (int r = 0; r < 8; ++r) {
            float mx = fmaxf(s0[r], s1[r]);
#pragma unroll
            for (int dmask = 8; dmask >= 1; dmask >>= 1)
                mx = fmaxf(mx, __shfl_xor(mx, dmask, 32));
            const float mnew  = fmaxf(m_run[r], mx);
            const float scale = __expf(m_run[r] - mnew);
            const float p0 = __expf(s0[r] - mnew);
            const float p1 = __expf(s1[r] - mnew);
            float rs = p0 + p1;
#pragma unroll
            for (int dmask = 8; dmask >= 1; dmask >>= 1)
                rs += __shfl_xor(rs, dmask, 32);
            l_run[r] = l_run[r] * scale + rs;
            m_run[r] = mnew;
#pragma unroll
            for (int t = 0; t < 4; ++t) acc[t][r] *= scale;
            myP[(r + 8 * rr) * 32 + cc]      = (_Float16)p0;
            myP[(r + 8 * rr) * 32 + 16 + cc] = (_Float16)p1;
        }

        // ---- O += P * V (P via LDS relayout, V fragments from LDS) ----
        const v16h ap = load_a(myP, 32, lane);
#pragma unroll
        for (int t = 0; t < 4; ++t) {
            const v16h bv = load_b(Vb + (t * 16) * 32, 32, lane);
            acc[t] = wmma16(ap, bv, acc[t]);
        }

        __syncthreads();   // tile fully consumed before buffer reuse (WAR guard)
    }

    // ---- normalize + write ctx [B,SQ,H*DH] ----
#pragma unroll
    for (int t = 0; t < 4; ++t)
#pragma unroll
        for (int r = 0; r < 8; ++r) {
            const int q = q0 + r + 8 * rr;
            const float v = acc[t][r] / l_run[r];
            ctx[(((size_t)b * SQ_ + q) * H_ + h) * DH_ + t * 16 + cc] = (_Float16)v;
        }
}

// ---------------------------------------------------------------------------
// Host launcher
// ---------------------------------------------------------------------------
extern "C" void kernel_launch(void* const* d_in, const int* in_sizes, int n_in,
                              void* d_out, int out_size, void* d_ws, size_t ws_size,
                              hipStream_t stream) {
    (void)in_sizes; (void)n_in; (void)out_size; (void)ws_size;
    const float* x   = (const float*)d_in[0];  // [B,SQ,D]
    const float* enc = (const float*)d_in[1];  // [B,SK,D]
    const float* pb  = (const float*)d_in[2];  // [1,H,SQ,SK]
    const float* Wq  = (const float*)d_in[3];  // [D, H*DH]
    const float* Wk  = (const float*)d_in[4];
    const float* Wv  = (const float*)d_in[5];
    const float* Wo  = (const float*)d_in[6];  // [H*DH, D]

    char* ws = (char*)d_ws;
    const size_t MB = 1u << 20;
    _Float16* xb  = (_Float16*)(ws + 0 * MB);   // [B*SQ, D]      8MB
    _Float16* eb  = (_Float16*)(ws + 8 * MB);   // [B*SK, D]      8MB
    _Float16* wqt = (_Float16*)(ws + 16 * MB);  // [N,K]          2MB
    _Float16* wkt = (_Float16*)(ws + 18 * MB);
    _Float16* wvt = (_Float16*)(ws + 20 * MB);
    _Float16* wot = (_Float16*)(ws + 22 * MB);
    _Float16* qd  = (_Float16*)(ws + 24 * MB);  // [B,H,SQ,DH]    8MB
    _Float16* kd  = (_Float16*)(ws + 32 * MB);  // [B,H,SK,DH]    8MB
    _Float16* vtd = (_Float16*)(ws + 40 * MB);  // [B,H,DH,SK]    8MB
    _Float16* ctx = (_Float16*)(ws + 48 * MB);  // [B,SQ,H*DH]    8MB

    const int nAct4 = (B_ * SQ_ * D_) / 4;  // 1M float4
    cvt_f32_f16<<<(nAct4 + 255) / 256, 256, 0, stream>>>(x, xb, nAct4);
    cvt_f32_f16<<<(nAct4 + 255) / 256, 256, 0, stream>>>(enc, eb, nAct4);

    dim3 tg(D_ / 64, D_ / 64);  // 16x16
    transpose_cvt<<<tg, 256, 0, stream>>>(Wq, wqt, D_, H_ * DH_);
    transpose_cvt<<<tg, 256, 0, stream>>>(Wk, wkt, D_, H_ * DH_);
    transpose_cvt<<<tg, 256, 0, stream>>>(Wv, wvt, D_, H_ * DH_);
    transpose_cvt<<<tg, 256, 0, stream>>>(Wo, wot, H_ * DH_, D_);

    dim3 gg((H_ * DH_) / 128, (B_ * SQ_) / 128);  // (8, 32)
    gemm_f16<0><<<gg, 128, 0, stream>>>(xb, wqt, qd,  B_ * SQ_, H_ * DH_, D_);
    gemm_f16<0><<<gg, 128, 0, stream>>>(eb, wkt, kd,  B_ * SK_, H_ * DH_, D_);
    gemm_f16<2><<<gg, 128, 0, stream>>>(eb, wvt, vtd, B_ * SK_, H_ * DH_, D_);

    attn_kernel<<<B_ * H_ * (SQ_ / 64), 128, 0, stream>>>(qd, kd, vtd, pb, ctx);

    dim3 go(D_ / 128, (B_ * SQ_) / 128);
    gemm_f16<1><<<go, 128, 0, stream>>>(ctx, wot, d_out, B_ * SQ_, D_, H_ * DH_);
}
